// SOGC_49778670961395
// MI455X (gfx1250) — compile-verified, matrix operand order checked by reading, not run on hardware
//
#include <hip/hip_runtime.h>
#include <hip/hip_bf16.h>

#define E_N    62
#define IN_FD  512
#define BN_FD  64
#define OUT_FD 256
#define TOPK   10

typedef __attribute__((ext_vector_type(16))) __bf16 v16bf;
typedef __attribute__((ext_vector_type(8)))  __bf16 v8bf;
typedef __attribute__((ext_vector_type(8)))  float  v8f;

// LDS row strides (elements). bf16 ones multiples of 8, f32 of 4, so every row is 16B-aligned
// for b128 fragment loads; odd-ish bank phases to avoid 16-way conflicts.
#define H_STRIDE  520   // bf16: 512 cols + 8 pad
#define G_STRIDE  80
#define A_STRIDE  68    // f32 scores
#define AB_STRIDE 80
#define PT_STRIDE 80

union V16U { v16bf v; v8bf h[2]; };

// Loads one 16x32 bf16 WMMA operand (A, or B given as "rows = columns of B",
// i.e. the weight matrix stored row-major is exactly B^T) from a row-major
// tile. Per ISA 7.12.2: lanes 0-15 hold row (row16+lane), K = k0+{0..7,16..23};
// lanes 16-31 hold row (row16+lane-16), K = k0+{8..15,24..31}.
__device__ __forceinline__ v16bf load_frag_row16(const __bf16* base, int stride,
                                                 int row16, int k0, int lane) {
  const int l  = lane & 15;
  const int hi = lane >> 4;
  const __bf16* p = base + (row16 + l) * stride + k0 + hi * 8;
  V16U u;
  u.h[0] = *(const v8bf*)(p);        // K = k0+hi*8 .. +7
  u.h[1] = *(const v8bf*)(p + 16);   // K = k0+hi*8+16 .. +23
  return u.v;
}

// f32 -> bf16 weight conversion into workspace:
// ws[0 .. 64*512)          = bn_w as bf16
// ws[64*512 .. +256*512)   = gc_w as bf16
__global__ void sogc_prep(const float* __restrict__ bn_w,
                          const float* __restrict__ gc_w, void* wraw) {
  __bf16* wbf = (__bf16*)wraw;
  const int i = blockIdx.x * blockDim.x + threadIdx.x;
  if (i < BN_FD * IN_FD)  wbf[i]                  = (__bf16)bn_w[i];
  if (i < OUT_FD * IN_FD) wbf[BN_FD * IN_FD + i]  = (__bf16)gc_w[i];
}

__global__ __launch_bounds__(256) void sogc_main(const float* __restrict__ x,
                                                 const float* __restrict__ bn_b,
                                                 const float* __restrict__ gc_b,
                                                 const void* __restrict__ wraw,
                                                 float* __restrict__ out) {
  __shared__ __align__(16) __bf16 h_s [64 * H_STRIDE];    //  65.0 KB  h (bf16)
  __shared__ __align__(16) __bf16 g_s [64 * G_STRIDE];    //  10.0 KB  tanh bottleneck
  __shared__ __align__(16) float  a_s [64 * A_STRIDE];    //  17.0 KB  raw scores
  __shared__ __align__(16) __bf16 ab_s[64 * AB_STRIDE];   //  10.0 KB  sparse attention
  __shared__ __align__(16) __bf16 pt_s[OUT_FD * PT_STRIDE]; // 40.0 KB  (h@gc_w^T)^T

  const int b    = blockIdx.x;
  const int tid  = threadIdx.x;
  const int wave = tid >> 5;
  const int lane = tid & 31;

  const __bf16* bnw = (const __bf16*)wraw;            // [64][512]
  const __bf16* gcw = bnw + BN_FD * IN_FD;            // [256][512]

  // Warm L2 for the next batch item's tile (global_prefetch_b8).
  if (b + 1 < (int)gridDim.x)
    __builtin_prefetch(x + (size_t)(b + 1) * E_N * IN_FD + tid * 64, 0, 1);

  // ---- Stage 1: stage x[b] (62x512 f32) into LDS as bf16; zero pad rows ----
  const float* xb = x + (size_t)b * E_N * IN_FD;
  for (int idx = tid * 4; idx < E_N * IN_FD; idx += 256 * 4) {
    const float4 v = *(const float4*)(xb + idx);
    const int r = idx >> 9;          // / 512
    const int c = idx & (IN_FD - 1); // % 512
    __bf16* d = h_s + r * H_STRIDE + c;
    d[0] = (__bf16)v.x; d[1] = (__bf16)v.y; d[2] = (__bf16)v.z; d[3] = (__bf16)v.w;
  }
  for (int idx = tid; idx < 2 * H_STRIDE; idx += 256)
    h_s[62 * H_STRIDE + idx] = (__bf16)0.0f;
  __syncthreads();

  // ---- Stage 2: g = tanh(h @ bn_w^T + bn_b)   [64x64], K=512 ----
  for (int t = wave; t < 16; t += 8) {
    const int tm = t >> 2, tn = t & 3;
    v8f acc = {};
    for (int k0 = 0; k0 < IN_FD; k0 += 32) {
      v16bf af = load_frag_row16(h_s, H_STRIDE, tm * 16, k0, lane);
      v16bf bf = load_frag_row16(bnw, IN_FD,    tn * 16, k0, lane);
      acc = __builtin_amdgcn_wmma_f32_16x16x32_bf16(false, af, false, bf,
                                                    (short)0, acc, false, false);
    }
    const int n     = tn * 16 + (lane & 15);
    const float bia = bn_b[n];
    const int mbase = tm * 16 + ((lane >> 4) << 3);
#pragma unroll
    for (int r = 0; r < 8; ++r)
      g_s[(mbase + r) * G_STRIDE + n] = (__bf16)tanhf(acc[r] + bia);
  }
  __syncthreads();

  // ---- Stage 3: scores s = g @ g^T   [64x64], K=64 (B cols = g rows) ----
  for (int t = wave; t < 16; t += 8) {
    const int tm = t >> 2, tn = t & 3;
    v8f acc = {};
    for (int k0 = 0; k0 < BN_FD; k0 += 32) {
      v16bf af = load_frag_row16(g_s, G_STRIDE, tm * 16, k0, lane);
      v16bf bf = load_frag_row16(g_s, G_STRIDE, tn * 16, k0, lane);
      acc = __builtin_amdgcn_wmma_f32_16x16x32_bf16(false, af, false, bf,
                                                    (short)0, acc, false, false);
    }
    const int n     = tn * 16 + (lane & 15);
    const int mbase = tm * 16 + ((lane >> 4) << 3);
#pragma unroll
    for (int r = 0; r < 8; ++r)
      a_s[(mbase + r) * A_STRIDE + n] = acc[r];
  }
  __syncthreads();

  // ---- Stage 4: per-row softmax over 62, top-10 threshold, unit diagonal ----
  for (int row = wave; row < 64; row += 8) {
    if (row >= E_N) {
      for (int c = lane; c < 64; c += 32) ab_s[row * AB_STRIDE + c] = (__bf16)0.0f;
      continue;
    }
    const int c0 = lane, c1 = lane + 32;
    float s0 = a_s[row * A_STRIDE + c0];
    float s1 = (c1 < E_N) ? a_s[row * A_STRIDE + c1] : -3.0e38f;
    float mx = fmaxf(s0, s1);
    for (int o = 16; o; o >>= 1) mx = fmaxf(mx, __shfl_xor(mx, o, 32));
    float e0 = __expf(s0 - mx);
    float e1 = (c1 < E_N) ? __expf(s1 - mx) : 0.0f;
    float sm = e0 + e1;
    for (int o = 16; o; o >>= 1) sm += __shfl_xor(sm, o, 32);
    const float inv = 1.0f / sm;
    const float p0 = e0 * inv;
    const float p1 = e1 * inv;
    // 10th-largest via iterative bounded max (probs in [0,1], -1 sentinel).
    float v1c = (c1 < E_N) ? p1 : -1.0f;
    float thr = 3.0e38f;
    for (int k = 0; k < TOPK; ++k) {
      float m = fmaxf((p0 < thr) ? p0 : -1.0f, (v1c < thr) ? v1c : -1.0f);
      for (int o = 16; o; o >>= 1) m = fmaxf(m, __shfl_xor(m, o, 32));
      thr = m;
    }
    ab_s[row * AB_STRIDE + c0] =
        (c0 == row) ? (__bf16)1.0f : ((p0 >= thr) ? (__bf16)p0 : (__bf16)0.0f);
    __bf16 o1 = (__bf16)0.0f;
    if (c1 < E_N) o1 = (c1 == row) ? (__bf16)1.0f
                                   : ((p1 >= thr) ? (__bf16)p1 : (__bf16)0.0f);
    ab_s[row * AB_STRIDE + c1] = o1;
  }
  __syncthreads();

  // ---- Stage 5: p = h @ gc_w^T  [64x256], K=512; store transposed -> pt_s ----
  for (int t = wave; t < 64; t += 8) {
    const int tm = t >> 4, tn = t & 15;
    v8f acc = {};
    for (int k0 = 0; k0 < IN_FD; k0 += 32) {
      v16bf af = load_frag_row16(h_s, H_STRIDE, tm * 16, k0, lane);
      v16bf bf = load_frag_row16(gcw, IN_FD,    tn * 16, k0, lane);
      acc = __builtin_amdgcn_wmma_f32_16x16x32_bf16(false, af, false, bf,
                                                    (short)0, acc, false, false);
    }
    const int n     = tn * 16 + (lane & 15);
    const int mbase = tm * 16 + ((lane >> 4) << 3);
#pragma unroll
    for (int r = 0; r < 8; ++r)
      pt_s[n * PT_STRIDE + (mbase + r)] = (__bf16)acc[r];   // transpose on store
  }
  __syncthreads();

  // ---- Stage 6: out = relu(a_sparse @ p + gc_b)  [62x256], K=64 ----
  float* ob = out + (size_t)b * E_N * OUT_FD;
  for (int t = wave; t < 64; t += 8) {
    const int tm = t >> 4, tn = t & 15;
    v8f acc = {};
    for (int k0 = 0; k0 < 64; k0 += 32) {
      v16bf af = load_frag_row16(ab_s, AB_STRIDE, tm * 16, k0, lane);
      v16bf bf = load_frag_row16(pt_s, PT_STRIDE, tn * 16, k0, lane);
      acc = __builtin_amdgcn_wmma_f32_16x16x32_bf16(false, af, false, bf,
                                                    (short)0, acc, false, false);
    }
    const int n     = tn * 16 + (lane & 15);
    const float bia = gc_b[n];
    const int mbase = tm * 16 + ((lane >> 4) << 3);
#pragma unroll
    for (int r = 0; r < 8; ++r) {
      const int m = mbase + r;
      if (m < E_N)
        ob[(size_t)m * OUT_FD + n] = fmaxf(acc[r] + bia, 0.0f);
    }
  }
}

extern "C" void kernel_launch(void* const* d_in, const int* in_sizes, int n_in,
                              void* d_out, int out_size, void* d_ws, size_t ws_size,
                              hipStream_t stream) {
  const float* x    = (const float*)d_in[0];
  const float* bn_w = (const float*)d_in[1];
  const float* bn_b = (const float*)d_in[2];
  const float* gc_w = (const float*)d_in[3];
  const float* gc_b = (const float*)d_in[4];
  float* out = (float*)d_out;

  const int batches = (in_sizes[0] / IN_FD) / E_N;   // 2048

  sogc_prep<<<(OUT_FD * IN_FD + 255) / 256, 256, 0, stream>>>(bn_w, gc_w, d_ws);
  sogc_main<<<batches, 256, 0, stream>>>(x, bn_b, gc_b, d_ws, out);
}
